// HausdorffLoss_26018911879200
// MI455X (gfx1250) — compile-verified, matrix operand order checked by reading, not run on hardware
//
#include <hip/hip_runtime.h>
#include <hip/hip_bf16.h>
#include <stdint.h>

typedef __attribute__((ext_vector_type(8))) int v8i;

#define H 512
#define W 512
#define IMG_ELEMS (H * W)

__device__ __forceinline__ int clamp511(int k) {
    k = k < 0 ? 0 : k;
    return k > 511 ? 511 : k;
}

// ---------------------------------------------------------------------------
// K0: zero the per-(img,batch) count and max-distance accumulators.
// ---------------------------------------------------------------------------
__global__ void k0_zero(int* counts, unsigned* hmax) {
    int t = threadIdx.x;
    if (t < 16) { counts[t] = 0; hmax[t] = 0u; }
}

// ---------------------------------------------------------------------------
// K1: mask "any true" counts via V_WMMA_I32_16X16X64_IU8.
// With B = all-ones bytes, the sum of all 256 C entries equals
// 16 * popcount(A bytes), independent of the A-operand layout, so packing
// order is irrelevant (only zero / non-zero is consumed downstream).
// grid = 16 blocks (img*8+batch), block = 256 threads (8 waves).
// ---------------------------------------------------------------------------
__global__ void k1_count_wmma(const float* __restrict__ inA,
                              const float* __restrict__ inB,
                              int* __restrict__ counts) {
    int ib    = blockIdx.x;          // img*8 + batch
    int img   = ib >> 3;
    int batch = ib & 7;
    const float* src = (img ? inB : inA) + (size_t)batch * IMG_ELEMS;

    int wave = threadIdx.x >> 5;     // 0..7
    int lane = threadIdx.x & 31;

    v8i c = {0, 0, 0, 0, 0, 0, 0, 0};
    v8i ones;
#pragma unroll
    for (int d = 0; d < 8; ++d) ones[d] = 0x01010101;

    int base = wave * (IMG_ELEMS / 8);       // 32768 floats per wave
    for (int it = 0; it < 32; ++it) {        // 1024 floats per WMMA
        int seg = base + it * 1024;
        v8i a;
#pragma unroll
        for (int d = 0; d < 8; ++d) {
            unsigned u = 0;
#pragma unroll
            for (int b = 0; b < 4; ++b) {
                float x = src[seg + (d * 4 + b) * 32 + lane];   // coalesced
                u |= (x > 0.5f ? 1u : 0u) << (8 * b);
            }
            a[d] = (int)u;
        }
        c = __builtin_amdgcn_wmma_i32_16x16x64_iu8(false, a, false, ones, c,
                                                   false, false);
    }
    int s = c[0] + c[1] + c[2] + c[3] + c[4] + c[5] + c[6] + c[7];
#pragma unroll
    for (int off = 16; off; off >>= 1) s += __shfl_xor(s, off, 32);
    if (lane == 0) atomicAdd(&counts[ib], s);   // 16 * popcount; only !=0 used
}

// ---------------------------------------------------------------------------
// K2: axis-0 (column) 1-D distance, clamped to 2048, stored as u16.
// Clamp is safe: any non-empty mask yields a 2-D candidate <= 2*511^2
// = 522242 < 2048^2; the clamped sentinel can only win when the whole mask
// is empty, which the empty path (K1 counts) overrides.
// One lane per (img,batch,col): 8192 lanes, coalesced row sweeps.
// ---------------------------------------------------------------------------
__global__ void k2_edt_axis0(const float* __restrict__ inA,
                             const float* __restrict__ inB,
                             unsigned short* __restrict__ d1) {
    int t     = blockIdx.x * blockDim.x + threadIdx.x;   // 0..8191
    int img   = t >> 12;
    int rem   = t & 4095;
    int batch = rem >> 9;
    int col   = rem & 511;

    const float* src = (img ? inB : inA) + (size_t)batch * IMG_ELEMS;
    unsigned short* dcol = d1 + (size_t)(img * 8 + batch) * IMG_ELEMS;

    int pos = -100000;                       // nearest true at index <= i
    for (int i = 0; i < H; ++i) {
        float x = src[i * W + col];
        if (x > 0.5f) pos = i;
        int d = i - pos; if (d > 2048) d = 2048;
        dcol[i * W + col] = (unsigned short)d;
    }
    pos = 100000 + H;                        // nearest true at index >= i
    for (int i = H - 1; i >= 0; --i) {
        float x = src[i * W + col];
        if (x > 0.5f) pos = i;
        int du = pos - i; if (du > 2048) du = 2048;
        int dd = (int)dcol[i * W + col];
        int d  = du < dd ? du : dd;
        dcol[i * W + col] = (unsigned short)d;
    }
}

// ---------------------------------------------------------------------------
// K3: lower envelope along axis 1 + masked-max Hausdorff accumulation.
// D[i,j] = min_k ((j-k)^2 + f[i,k]) = j^2 + min_k (g[k] - 2jk),
// with g[k] = f[k] + k^2 staged in LDS (row fetched via the CDNA5 async
// global->LDS path / ASYNCcnt).
// Exact pruning: the optimal k satisfies |j-k| <= d1[i,j], so scanning the
// window of half-width R = wave-max(d1) around j is exact; clamped extra
// iterations only add valid candidates. Fallback to the full scan if R>=256.
// All candidate values are integers < 2^24, so fp32 arithmetic is exact and
// matches the reference bit-for-bit.
// grid = (512 rows, 8 batches, 2 images), block = 512 threads.
// ---------------------------------------------------------------------------
__global__ void k3_lower_env(const float* __restrict__ inA,
                             const float* __restrict__ inB,
                             const unsigned short* __restrict__ d1,
                             unsigned* __restrict__ hmax) {
    int row = blockIdx.x, batch = blockIdx.y, img = blockIdx.z;
    __shared__ unsigned            rawbuf[W / 2];     // 512 u16 as 256 dwords
    __shared__ __align__(16) float g[W];
    __shared__ float               redbuf[16];

    const unsigned short* drow =
        d1 + ((size_t)(img * 8 + batch) * H + row) * W;
    int j = threadIdx.x;

    // --- async copy of the d1 row into LDS (256 x b32) ---
    if (j < 256) {
        unsigned ldsOff = (unsigned)(size_t)(&rawbuf[0]) + (unsigned)j * 4u;
        unsigned gOff   = (unsigned)j * 4u;
        unsigned long long gbase = (unsigned long long)(size_t)drow;
        asm volatile("global_load_async_to_lds_b32 %0, %1, %2"
                     :: "v"(ldsOff), "v"(gOff), "s"(gbase)
                     : "memory");
        asm volatile("s_wait_asynccnt 0" ::: "memory");
    }
    __syncthreads();

    unsigned pr = rawbuf[j >> 1];
    int dv = (int)((j & 1) ? (pr >> 16) : (pr & 0xFFFFu));
    {   // g[j] = f[j] + j^2  (exact integers in fp32, all < 2^24)
        float df = (float)dv;
        float jf = (float)j;
        g[j] = df * df + jf * jf;
    }
    __syncthreads();

    // wave-uniform exact scan radius
    int R = dv;
#pragma unroll
    for (int off = 16; off; off >>= 1) R = max(R, __shfl_xor(R, off, 32));

    float n2j = -2.0f * (float)j;
    float m0 = 3.0e38f, m1 = 3.0e38f, m2 = 3.0e38f, m3 = 3.0e38f;

    if (R < 256) {
        // pruned window scan: 2R+1 candidates (padded to x4, clamped pads
        // are still valid candidates -> exact)
        int base = j - R;
        int cnt  = 2 * R + 1;
        for (int t = 0; t < cnt; t += 4) {
            int k0 = clamp511(base + t);
            int k1 = clamp511(base + t + 1);
            int k2 = clamp511(base + t + 2);
            int k3 = clamp511(base + t + 3);
            m0 = fminf(m0, fmaf(n2j, (float)k0, g[k0]));
            m1 = fminf(m1, fmaf(n2j, (float)k1, g[k1]));
            m2 = fminf(m2, fmaf(n2j, (float)k2, g[k2]));
            m3 = fminf(m3, fmaf(n2j, (float)k3, g[k3]));
        }
    } else {
        // full scan, float4 LDS broadcasts, 4 independent min chains
        float kf = 0.0f;
#pragma unroll 4
        for (int k = 0; k < W; k += 4) {
            float4 gg = *reinterpret_cast<const float4*>(&g[k]);
            m0 = fminf(m0, fmaf(n2j, kf,        gg.x));
            m1 = fminf(m1, fmaf(n2j, kf + 1.0f, gg.y));
            m2 = fminf(m2, fmaf(n2j, kf + 2.0f, gg.z));
            m3 = fminf(m3, fmaf(n2j, kf + 3.0f, gg.w));
            kf += 4.0f;
        }
    }
    float m  = fminf(fminf(m0, m1), fminf(m2, m3));
    float D  = m + (float)j * (float)j;
    float dt = sqrtf(D);

    // img==0 -> EDT of a(inputs): mask with b(targets); img==1 -> mask with a.
    const float* other = (img ? inA : inB) + (size_t)batch * IMG_ELEMS;
    float x   = other[row * W + j];
    float val = (x > 0.5f) ? dt : 0.0f;

#pragma unroll
    for (int off = 16; off; off >>= 1)
        val = fmaxf(val, __shfl_xor(val, off, 32));
    int wave = j >> 5, lane = j & 31;
    if (lane == 0) redbuf[wave] = val;
    __syncthreads();
    if (wave == 0) {
        float v = (lane < 16) ? redbuf[lane] : 0.0f;
#pragma unroll
        for (int off = 8; off; off >>= 1)
            v = fmaxf(v, __shfl_xor(v, off, 32));
        if (lane == 0)
            atomicMax(&hmax[img * 8 + batch], __float_as_uint(v));
    }
}

// ---------------------------------------------------------------------------
// K4: combine: hd = max over both directions; empty -> 1 else 1 - 1/(1+hd).
// ---------------------------------------------------------------------------
__global__ void k4_finalize(const int* __restrict__ counts,
                            const unsigned* __restrict__ hmax,
                            float* __restrict__ out) {
    if (threadIdx.x == 0 && blockIdx.x == 0) {
        float s = 0.0f;
        for (int b = 0; b < 8; ++b) {
            bool empty = (counts[b] == 0) || (counts[8 + b] == 0);
            float h = fmaxf(__uint_as_float(hmax[b]),
                            __uint_as_float(hmax[8 + b]));
            float v = empty ? 1.0f : (1.0f - 1.0f / (1.0f + h));
            s += v;
        }
        out[0] = s * 0.125f;
    }
}

// ---------------------------------------------------------------------------
// Launch. Workspace layout:
//   [0,64)    : int   counts[16]   (img*8+batch, 16*popcount of mask)
//   [64,128)  : uint  hmax[16]     (float bits, non-negative => int-orderable)
//   [128, ..) : u16   d1[2*8*512*512]  = 8 MiB axis-0 distances
// ---------------------------------------------------------------------------
extern "C" void kernel_launch(void* const* d_in, const int* in_sizes, int n_in,
                              void* d_out, int out_size, void* d_ws,
                              size_t ws_size, hipStream_t stream) {
    const float* inputs  = (const float*)d_in[0];
    const float* targets = (const float*)d_in[1];
    float* out = (float*)d_out;

    int*            counts = (int*)d_ws;
    unsigned*       hmax   = (unsigned*)((char*)d_ws + 64);
    unsigned short* d1     = (unsigned short*)((char*)d_ws + 128);

    k0_zero<<<1, 32, 0, stream>>>(counts, hmax);
    k1_count_wmma<<<16, 256, 0, stream>>>(inputs, targets, counts);
    k2_edt_axis0<<<32, 256, 0, stream>>>(inputs, targets, d1);
    k3_lower_env<<<dim3(512, 8, 2), 512, 0, stream>>>(inputs, targets, d1,
                                                      hmax);
    k4_finalize<<<1, 32, 0, stream>>>(counts, hmax, out);
}
